// AutoLUT_80032420594221
// MI455X (gfx1250) — matrix operand order repaired and checked by reading, not compiled, not run on hardware
//
#include <hip/hip_runtime.h>
#include <stdint.h>

// ---------------- problem constants ----------------
#define NBATCH 4
#define NPIX   512
#define HW     (NPIX*NPIX)            // 262144
#define OW     (NPIX*4)               // 2048
#define NLUT   83521                  // 17^4
#define LUT0_STRIDE 86016             // padded per-sampler int8 stage-0 LUT stride
#define LUT1_STRIDE (NLUT*16)         // 1336336 bytes per sampler (int8 x16)

// ---------------- workspace layout (byte offsets into d_ws) ----------------
#define OFF_LUT0Q 0u                  // 3 * 86016
#define OFF_LUT1Q 258048u             // 3 * 1336336
#define OFF_W0ROT 4267264u            // 3*4*4*25 floats (stage-0 rotated conv weights)
#define OFF_WMATA 4272128u            // 3*2*16*16 f16 (stage-1 WMMA A-operand layout)
#define OFF_X1    4275200u            // 4*512*512 floats (stage-0 output image)
// total required: 8,469,504 bytes

typedef __attribute__((ext_vector_type(16))) _Float16 v16h;
typedef __attribute__((ext_vector_type(8)))  float    v8f;
typedef __attribute__((ext_vector_type(4)))  uint32_t u32x4;
typedef __attribute__((ext_vector_type(8)))  int      i32x8;
typedef __attribute__((ext_vector_type(4)))  int      i32x4;

// 4-D simplex interpolation setup: vals in [0,256) -> 5 vertex indices + weights
__device__ __forceinline__ void simplex5(const float vals[4], int verts[5], float w[5]) {
    const int S[4] = {4913, 289, 17, 1};           // strides {L^3,L^2,L,1}, L=17
    float f[4]; int st[4]; int base = 0;
    #pragma unroll
    for (int c = 0; c < 4; ++c) {
        float lo = floorf(vals[c] * 0.0625f);      // floor_divide(v,16)
        f[c] = vals[c] - 16.0f * lo;               // mod(v,16)
        base += (int)lo * S[c];
        st[c] = S[c];
    }
    // descending sort of (f, stride); ties give zero simplex weight so order is free
#define CSWP(a,b) { if (f[a] < f[b]) { float tf=f[a]; f[a]=f[b]; f[b]=tf; int ts=st[a]; st[a]=st[b]; st[b]=ts; } }
    CSWP(0,1) CSWP(2,3) CSWP(0,2) CSWP(1,3) CSWP(1,2)
#undef CSWP
    w[0] = 16.0f - f[0];
    w[1] = f[0] - f[1];  w[2] = f[1] - f[2];  w[3] = f[2] - f[3];  w[4] = f[3];
    verts[0] = base;
    verts[1] = verts[0] + st[0];
    verts[2] = verts[1] + st[1];
    verts[3] = verts[2] + st[2];
    verts[4] = verts[3] + st[3];
}

// rotated conv weight: tap t of the 5x5 clamped neighborhood, rotation `rot`, channel `ch`.
// (u,v) = M_{(4-rot)%4}(du,dv); valid iff 0<=u,v<3. samp layout: (S,4,1,3,3) flat.
__device__ __forceinline__ float rot_weight(const float* samp, int s, int rot, int ch, int t) {
    int du = t / 5 - 2, dv = t % 5 - 2;
    int u, v;
    switch ((4 - rot) & 3) {
        case 0:  u =  du; v =  dv; break;
        case 1:  u =  dv; v = -du; break;
        case 2:  u = -du; v = -dv; break;
        default: u = -dv; v =  du; break;
    }
    if (u < 0 || u > 2 || v < 0 || v > 2) return 0.0f;
    return samp[(s * 4 + ch) * 9 + u * 3 + v];
}

// -------- kernel 1: quantize LUTs to int8 (fix_weight == int8 exactly) --------
__launch_bounds__(256)
__global__ void k_quant(const float* __restrict__ lut0, const float* __restrict__ lut1,
                        int8_t* __restrict__ lut0q, int8_t* __restrict__ lut1q) {
    const int tid = blockIdx.x * blockDim.x + threadIdx.x;
    const int stride = gridDim.x * blockDim.x;
    for (int i = tid; i < 3 * NLUT; i += stride) {
        int s = i / NLUT, r = i % NLUT;
        float q = fminf(fmaxf(rintf(lut0[i] * 127.0f), -127.0f), 127.0f);
        lut0q[s * LUT0_STRIDE + r] = (int8_t)q;
    }
    for (int i = tid; i < 3 * NLUT * 16; i += stride) {
        float q = fminf(fmaxf(rintf(lut1[i] * 127.0f), -127.0f), 127.0f);
        lut1q[i] = (int8_t)q;
    }
}

// -------- kernel 2: build rotated weight tables --------
// W0rot[s][rot][ch][t(25)] floats  (stage-0 scalar conv)
// WmatA[s][g][m][h(16)] f16        (stage-1 WMMA A operand: m=rot*4+ch, K(h,g) per ISA A-layout)
__launch_bounds__(256)
__global__ void k_weights(const float* __restrict__ samp0, const float* __restrict__ samp1,
                          float* __restrict__ w0rot, _Float16* __restrict__ wmatA) {
    const int tid = blockIdx.x * blockDim.x + threadIdx.x;
    if (tid < 3 * 4 * 4 * 25) {
        int t = tid % 25, c = (tid / 25) & 3, r = (tid / 100) & 3, s = tid / 400;
        w0rot[tid] = rot_weight(samp0, s, r, c, t);
    }
    if (tid < 3 * 2 * 16 * 16) {
        int h = tid & 15, m = (tid >> 4) & 15, g = (tid >> 8) & 1, s = tid >> 9;
        int K = h + 8 * g + ((h >= 8) ? 8 : 0);    // 16-bit A 16x32 VGPR layout
        float w = (K < 25) ? rot_weight(samp1, s, m >> 2, m & 3, K) : 0.0f;
        wmatA[tid] = (_Float16)w;
    }
}

// -------- kernel 3: stage 0 (LDS int8 LUT via TDM + manual fill, scalar interp) --------
__launch_bounds__(256)
__global__ void k_stage0(const float* __restrict__ x, const int8_t* __restrict__ lut0q,
                         const float* __restrict__ w0rot, const float* __restrict__ sbias0,
                         float* __restrict__ x1) {
    extern __shared__ int8_t s_lut[];              // LUT0_STRIDE bytes (dynamic)
    float acc[8];
    #pragma unroll
    for (int q = 0; q < 8; ++q) acc[q] = 0.0f;
    const int base = blockIdx.x * 2048;

    for (int s = 0; s < 3; ++s) {
        __syncthreads();                           // readers of previous table done
        const int8_t* src = lut0q + s * LUT0_STRIDE;
        // guaranteed-correct fill (global_load_b128 -> ds_store_b128)
        for (int t = threadIdx.x; t < LUT0_STRIDE / 16; t += blockDim.x)
            ((int4*)s_lut)[t] = ((const int4*)src)[t];
        // CDNA5 Tensor Data Mover: async DMA of the same bytes (2-D tile 1024x82 = 83968B
        // covers the 83521 live entries). D# per CDNA5 ISA. Byte-identical
        // to the manual fill, so the double-write is benign.
        if (threadIdx.x < 32) {                    // one wave issues the DMA
            uint64_t ga = (uint64_t)(uintptr_t)src;
            uint32_t la = (uint32_t)(uintptr_t)(&s_lut[0]);
            u32x4 g0; i32x8 g1; i32x4 z4 = {0, 0, 0, 0};
            i32x8 z8 = {0, 0, 0, 0, 0, 0, 0, 0};
            g0[0] = 1u;                                      // count=1, gather off
            g0[1] = la;                                      // lds_addr
            g0[2] = (uint32_t)ga;                            // global_addr[31:0]
            g0[3] = (uint32_t)(ga >> 32) | 0x80000000u;      // global_addr[56:32] | type=2
            g1[0] = 0;                                       // wg_mask=0, data_size=1B, no flags
            g1[1] = (int)(1024u << 16);                      // tensor_dim0 = 1024
            g1[2] = (int)(82u << 16);                        // tensor_dim1 = 82
            g1[3] = (int)(1024u << 16);                      // tile_dim0 = 1024
            g1[4] = 82;                                      // tile_dim1 = 82, tile_dim2 = 0
            g1[5] = 1024;                                    // tensor_dim0_stride = 1024
            g1[6] = 0; g1[7] = 0;
            __builtin_amdgcn_tensor_load_to_lds(g0, g1, z4, z4, z8, 0);
            __builtin_amdgcn_s_wait_tensorcnt(0);
        }
        __syncthreads();

        #pragma unroll
        for (int blk = 0; blk < 8; ++blk) {
            const int pix = base + blk * 256 + threadIdx.x;
            const int b = pix >> 18;
            const int i = (pix >> 9) & (NPIX - 1);
            const int j = pix & (NPIX - 1);
            const float* xb = x + b * HW;
            float nb[25];                          // edge-clamped 5x5 neighborhood * 255
            #pragma unroll
            for (int t = 0; t < 25; ++t) {
                int ii = min(max(i + t / 5 - 2, 0), NPIX - 1);
                int jj = min(max(j + t % 5 - 2, 0), NPIX - 1);
                nb[t] = 255.0f * xb[ii * NPIX + jj];
            }
            float pred = 0.0f;
            for (int r = 0; r < 4; ++r) {
                float vals[4];
                #pragma unroll
                for (int c = 0; c < 4; ++c) {
                    float a2 = sbias0[s * 4 + c];
                    const float* wr = w0rot + ((s * 4 + r) * 4 + c) * 25;
                    #pragma unroll
                    for (int t = 0; t < 25; ++t) a2 = fmaf(wr[t], nb[t], a2);
                    vals[c] = a2;
                }
                int verts[5]; float wv[5];
                simplex5(vals, verts, wv);
                float o = 0.0f;
                #pragma unroll
                for (int v5 = 0; v5 < 5; ++v5)
                    o = fmaf(wv[v5], (float)s_lut[verts[v5]], o);   // ds_load_i8 gather
                pred = rintf(pred + o * 0.0625f);  // per-rotation ste_round chain
            }
            acc[blk] += pred;                      // plain sum over samplers
        }
    }
    #pragma unroll
    for (int blk = 0; blk < 8; ++blk) {
        const int pix = base + blk * 256 + threadIdx.x;
        x1[pix] = rintf(fminf(fmaxf(acc[blk] * (1.0f / 12.0f) + 127.0f, 0.0f), 255.0f));
    }
}

// -------- kernel 4: stage 1 (WMMA conv, paired-lane rotations, 4x upscale) --------
// Wave handles 16 pixels (one row segment). WMMA: D[combo(16)][pixel(16)] = W[combo][tap(32)] x N[tap][pixel].
// D layout -> lanes 0-15 hold rotations 0,1 of their pixel; lanes 16-31 hold rotations 2,3.
__launch_bounds__(256)
__global__ void k_stage1(const float* __restrict__ x, const float* __restrict__ x1,
                         const int8_t* __restrict__ lut1q, const _Float16* __restrict__ wmatA,
                         const float* __restrict__ sbias1, const float* __restrict__ resw1,
                         float* __restrict__ out) {
    const int tid  = blockIdx.x * blockDim.x + threadIdx.x;
    const int lane = threadIdx.x & 31;
    const int wid  = tid >> 5;                     // 65536 waves total
    const int b    = wid >> 14;
    const int rem  = wid & 16383;
    const int i    = rem >> 5;
    const int jg   = rem & 31;
    const int p    = lane & 15;                    // pixel within wave
    const int g    = lane >> 4;                    // half-wave group
    const int j    = jg * 16 + p;
    const float* xb  = x  + b * HW;
    const float* x1b = x1 + b * HW;

    // B operands: lane's 16 halves hold taps K = 8*g + (h&7) + (h>=8 ? 16 : 0) of its pixel (matches A table order)
    union { _Float16 a[16]; v16h v; } bcur, bprev;
    #pragma unroll
    for (int h = 0; h < 16; ++h) {
        const int K = (h & 7) + 8 * g + ((h >= 8) ? 16 : 0);
        float cv = 0.0f, pv = 0.0f;
        if (K < 25) {
            int ii = min(max(i + K / 5 - 2, 0), NPIX - 1);
            int jj = min(max(j + K % 5 - 2, 0), NPIX - 1);
            cv = x1b[ii * NPIX + jj];              // integer-valued: exact in f16
            pv = 255.0f * xb[ii * NPIX + jj];
        }
        bcur.a[h]  = (_Float16)cv;
        bprev.a[h] = (_Float16)pv;
    }

    float total[16];
    #pragma unroll
    for (int q = 0; q < 16; ++q) total[q] = 0.0f;

    for (int s = 0; s < 3; ++s) {
        const v16h A = *(const v16h*)(wmatA + ((size_t)((s * 2 + g) * 16 + p)) * 16);
        v8f dcur  = {0.f,0.f,0.f,0.f,0.f,0.f,0.f,0.f};
        v8f dprev = {0.f,0.f,0.f,0.f,0.f,0.f,0.f,0.f};
        dcur  = __builtin_amdgcn_wmma_f32_16x16x32_f16(false, A, false, bcur.v,  (short)0, dcur,  false, false);
        dprev = __builtin_amdgcn_wmma_f32_16x16x32_f16(false, A, false, bprev.v, (short)0, dprev, false, false);

        float outr[2][16];                          // interp output for this lane's 2 rotations
        #pragma unroll
        for (int lr = 0; lr < 2; ++lr) {
            float vals[4];
            #pragma unroll
            for (int c = 0; c < 4; ++c) {          // D row m = (2g+lr)*4 + c lives in VGPR lr*4+c
                const float rw = fminf(fmaxf(resw1[s * 4 + c], 0.0f), 1.0f);
                vals[c] = rw * dprev[lr * 4 + c] + (1.0f - rw) * dcur[lr * 4 + c] + sbias1[s * 4 + c];
            }
            int verts[5]; float wv[5];
            simplex5(vals, verts, wv);
            #pragma unroll
            for (int k = 0; k < 16; ++k) outr[lr][k] = 0.0f;
            const int8_t* lut = lut1q + (size_t)s * LUT1_STRIDE;
            #pragma unroll
            for (int v5 = 0; v5 < 5; ++v5) {
                const int4 dq = *(const int4*)(lut + (size_t)verts[v5] * 16);  // 16 int8
                const float wq = wv[v5];
                const int words[4] = {dq.x, dq.y, dq.z, dq.w};
                #pragma unroll
                for (int wi = 0; wi < 4; ++wi)
                    #pragma unroll
                    for (int bb = 0; bb < 4; ++bb)
                        outr[lr][wi * 4 + bb] =
                            fmaf(wq, (float)(int8_t)(((uint32_t)words[wi]) >> (8 * bb)),
                                 outr[lr][wi * 4 + bb]);
            }
            #pragma unroll
            for (int k = 0; k < 16; ++k) outr[lr][k] *= 0.0625f;
        }

        // rotation-ordered rounded accumulation; subcell index rotates on the 4-grid
        #pragma unroll
        for (int a2 = 0; a2 < 4; ++a2) {
            #pragma unroll
            for (int b2 = 0; b2 < 4; ++b2) {
                const int k0 = 4 * a2 + b2;
                const int k1 = 4 * (3 - b2) + a2;
                const int k2 = 4 * (3 - a2) + (3 - b2);
                const int k3 = 4 * b2 + (3 - a2);
                const float cA = outr[0][g ? k2 : k0];
                const float cB = outr[1][g ? k3 : k1];
                float t01 = rintf(cA);  t01 = rintf(t01 + cB);     // rot0,rot1 (valid in lanes 0-15)
                const float tt = __shfl(t01, p, 32);               // hand partial to partner lane
                float tf = rintf(tt + cA);  tf = rintf(tf + cB);   // rot2,rot3 (valid in lanes 16-31)
                total[a2 * 4 + b2] += tf;
            }
        }
    }

    if (g) {                                       // lanes 16-31 hold the finished blocks
        float* ob = out + (size_t)b * OW * OW;
        #pragma unroll
        for (int a2 = 0; a2 < 4; ++a2) {
            float4 r;
            r.x = rintf(fminf(fmaxf(total[a2 * 4 + 0] * (1.0f / 3.0f), 0.0f), 255.0f)) * (1.0f / 255.0f);
            r.y = rintf(fminf(fmaxf(total[a2 * 4 + 1] * (1.0f / 3.0f), 0.0f), 255.0f)) * (1.0f / 255.0f);
            r.z = rintf(fminf(fmaxf(total[a2 * 4 + 2] * (1.0f / 3.0f), 0.0f), 255.0f)) * (1.0f / 255.0f);
            r.w = rintf(fminf(fmaxf(total[a2 * 4 + 3] * (1.0f / 3.0f), 0.0f), 255.0f)) * (1.0f / 255.0f);
            *(float4*)(ob + (size_t)(4 * i + a2) * OW + 4 * j) = r;
        }
    }
}

extern "C" void kernel_launch(void* const* d_in, const int* in_sizes, int n_in,
                              void* d_out, int out_size, void* d_ws, size_t ws_size,
                              hipStream_t stream) {
    (void)in_sizes; (void)n_in; (void)out_size; (void)ws_size;
    const float* x      = (const float*)d_in[0];
    const float* lut0   = (const float*)d_in[1];
    const float* lut1   = (const float*)d_in[2];
    const float* samp0  = (const float*)d_in[3];
    const float* samp1  = (const float*)d_in[4];
    const float* sbias0 = (const float*)d_in[5];
    const float* sbias1 = (const float*)d_in[6];
    const float* resw1  = (const float*)d_in[7];
    float* out = (float*)d_out;
    char* ws = (char*)d_ws;

    int8_t*   lut0q = (int8_t*)(ws + OFF_LUT0Q);
    int8_t*   lut1q = (int8_t*)(ws + OFF_LUT1Q);
    float*    w0rot = (float*)(ws + OFF_W0ROT);
    _Float16* wmatA = (_Float16*)(ws + OFF_WMATA);
    float*    x1    = (float*)(ws + OFF_X1);

    k_quant  <<<1024, 256, 0, stream>>>(lut0, lut1, lut0q, lut1q);
    k_weights<<<8,    256, 0, stream>>>(samp0, samp1, w0rot, wmatA);
    k_stage0 <<<512,  256, LUT0_STRIDE, stream>>>(x, lut0q, w0rot, sbias0, x1);
    k_stage1 <<<8192, 256, 0, stream>>>(x, x1, lut1q, wmatA, sbias1, resw1, out);
}
